// Mlpmoe_62491774157634
// MI455X (gfx1250) — compile-verified, matrix operand order checked by reading, not run on hardware
//
#include <hip/hip_runtime.h>
#include <hip/hip_bf16.h>
#include <cmath>

// CDNA5 / gfx1250, wave32. Matrix math via v_wmma_f32_16x16x32_bf16
// (f32 accumulate). Weights are pre-packed once per call into bf16
// WMMA-fragment order in workspace so the GEMM K-loops read B-fragments
// directly from global memory (coalesced b128 loads, no LDS staging, no
// barriers inside the K-loop). Each wave computes BOTH M-tiles per B
// fragment, halving global B traffic and giving two independent WMMA
// chains for XDL pipelining. mids input is unused by the reference.

typedef __attribute__((ext_vector_type(16))) __bf16 v16bf;
typedef __attribute__((ext_vector_type(8)))  __bf16 v8bf;
typedef __attribute__((ext_vector_type(4)))  __bf16 v4bf;
typedef __attribute__((ext_vector_type(8)))  float  v8f;

constexpr int kNcls = 6;
constexpr int kP    = 256;
constexpr int kD    = 768;
constexpr int kH    = 3072;
constexpr int kBM   = 32;           // rows per block (2 M-tiles)
constexpr int kK1S  = kD / 32;      // 24 K-steps for GEMM1
constexpr int kK2S  = kH / 32;      // 96 K-steps for GEMM2
constexpr long kWElems = (long)kD * kH;   // elements per weight matrix

// LDS: A fragments (2*24*512) + hidden fragments (2*96*512)
constexpr int kHOff      = 2 * kK1S * 512;            // 24576
constexpr int kSmemElems = kHOff + 2 * kK2S * 512;    // 122880 bf16 = 245760 B

struct RowMap { long base; long gstride; long rstride; long rpg; };

__device__ __forceinline__ long row_addr(const RowMap m, long r) {
  return m.base + (r / m.rpg) * m.gstride + (r % m.rpg) * m.rstride;
}

// ISA 7.12.2 16-bit A/B fragment mapping for 16x32 (K=32) tiles:
// lanes 0-15 hold K in {0..7, 16..23}, lanes 16-31 hold K in {8..15, 24..31}.
__device__ __forceinline__ int fl(int mn, int kk) { return mn + (((kk >> 3) & 1) << 4); }
__device__ __forceinline__ int fe(int kk)         { return (kk & 7) + ((kk >> 4) << 3); }

__device__ __forceinline__ float gelu_exact(float x) {
  return 0.5f * x * (1.0f + erff(x * 0.70710678118654752440f));
}

// ---------------------------------------------------------------------------
// Pack a K x N f32 row-major weight matrix into bf16 WMMA B-fragment order:
// dst[((ks*(N/16) + nt)*32 + lane)*16 + e], 512 bf16 per 32x16 (KxN) tile.
// Each thread produces 8 contiguous bf16 (one 16B store).
// ---------------------------------------------------------------------------
__global__ void pack_weights_kernel(const float* __restrict__ src,
                                    __bf16* __restrict__ dst,
                                    int K, int N)
{
  long o8 = (long)blockIdx.x * blockDim.x + threadIdx.x;
  long total = ((long)K * N) >> 3;
  if (o8 >= total) return;
  long o = o8 << 3;
  int e0      = (int)(o & 15);            // 0 or 8
  int lane    = (int)((o >> 4) & 31);
  long tile   = o >> 9;                   // ks * (N/16) + nt
  int ntiles  = N >> 4;
  int ks      = (int)(tile / ntiles);
  int nt      = (int)(tile - (long)ks * ntiles);
  int n       = nt * 16 + (lane & 15);
  int lane_hi = lane >> 4;
  int ehi     = e0 >> 3;
  v8bf out;
#pragma unroll
  for (int q = 0; q < 8; ++q) {
    int kk = q + 8 * lane_hi + 16 * ehi;  // inverse of fl/fe mapping
    out[q] = (__bf16)src[(long)(ks * 32 + kk) * N + n];
  }
  *reinterpret_cast<v8bf*>(dst + o) = out;
}

// ---------------------------------------------------------------------------
// Fused MLP: out = gelu(x @ W1 + b1) @ W2 + b2, hidden panel lives in LDS.
// W1f/W2f are bf16 fragment-packed. 8 waves = 8 N-slots; each wave computes
// both M-tiles per B fragment (one B load feeds two WMMAs).
// ---------------------------------------------------------------------------
__global__ void fused_mlp_kernel(
    const float* __restrict__ X,
    const __bf16* __restrict__ W1f, const float* __restrict__ B1,
    const __bf16* __restrict__ W2f, const float* __restrict__ B2,
    float* __restrict__ Out,
    RowMap in_map, RowMap out_map,
    int cls_mode,
    const __bf16* __restrict__ A1f, const float* __restrict__ A1b,
    const __bf16* __restrict__ A2f, const float* __restrict__ A2b)
{
  extern __shared__ __align__(32) __bf16 smem[];
  __bf16* Abuf = smem;
  __bf16* Hbuf = smem + kHOff;

  if (cls_mode) {
    // Expert slot e = class*2 + pair; PAIRS[i] = [(a,c), (c,a)] with
    // a = i/2, c = 3 + (i%2)  (derived from GT0).
    const int e = blockIdx.y;
    const int i = e >> 1, pair = e & 1;
    const int a = i >> 1, c = 3 + (i & 1);
    const int j = pair ? c : a;   // A1 expert
    const int k = pair ? a : c;   // A2 expert
    W1f = A1f + (long)j * kWElems;  B1 = A1b + (long)j * kH;
    W2f = A2f + (long)k * kWElems;  B2 = A2b + (long)k * kD;
    in_map.base  = (long)i * kD;
    out_map.base = (long)e * 64 * kD;
  }

  const int tid   = threadIdx.x;
  const int lane  = tid & 31;
  const int wave  = tid >> 5;         // wave = N-slot (0..7)
  const int lcol  = lane & 15;        // C/D fragment: N = lane%16
  const int lrow  = (lane >> 4) << 3; // C/D fragment: M = r + 8*(lane/16)
  const long row0 = (long)blockIdx.x * kBM;

  // ---- Stage A (input rows) once: float4 load -> 4x bf16 -> 8B LDS store ----
  for (int idx = tid; idx < kBM * (kD / 4); idx += 256) {
    int row = idx / (kD / 4);
    int col = (idx - row * (kD / 4)) * 4;             // aligned group of 4 K's
    const float4 v = *reinterpret_cast<const float4*>(
        X + row_addr(in_map, row0 + row) + col);
    int mtt = row >> 4, mm = row & 15;
    int ks = col >> 5, kk = col & 31;
    v4bf t;
    t[0] = (__bf16)v.x; t[1] = (__bf16)v.y; t[2] = (__bf16)v.z; t[3] = (__bf16)v.w;
    *reinterpret_cast<v4bf*>(
        Abuf + ((mtt * kK1S + ks) * 32 + fl(mm, kk)) * 16 + fe(kk)) = t;
  }
  __syncthreads();

  // ---- Phase 1: hidden = gelu(x @ W1 + b1) -> LDS bf16 A-fragments ----
  {
    const long bstep = (long)(kH >> 4) * 512;          // per-ks fragment stride
    for (int ch = 0; ch < kH / 128; ++ch) {            // 128-col chunks
      const int nt = ch * 8 + wave;
      const __bf16* bptr = W1f + (long)nt * 512 + lane * 16;
      v8f acc0 = {}, acc1 = {};                        // M-tile 0 / M-tile 1
#pragma unroll 4
      for (int ks = 0; ks < kK1S; ++ks) {
        v16bf b  = *reinterpret_cast<const v16bf*>(bptr + (long)ks * bstep);
        v16bf a0 = *reinterpret_cast<const v16bf*>(
            Abuf + (ks * 32 + lane) * 16);
        v16bf a1 = *reinterpret_cast<const v16bf*>(
            Abuf + ((kK1S + ks) * 32 + lane) * 16);
        acc0 = __builtin_amdgcn_wmma_f32_16x16x32_bf16(false, a0, false, b,
                                                       (short)0, acc0, false, false);
        acc1 = __builtin_amdgcn_wmma_f32_16x16x32_bf16(false, a1, false, b,
                                                       (short)0, acc1, false, false);
      }
      const int col = nt * 16 + lcol;                  // hidden column (= K of GEMM2)
      const float bias = B1[col];
      const int k2s = col >> 5, kk2 = col & 31;
#pragma unroll
      for (int r = 0; r < 8; ++r) {
        int mm = lrow + r;
        int slot = (fl(mm, kk2)) * 16 + fe(kk2);
        Hbuf[(0 * kK2S + k2s) * 512 + slot] = (__bf16)gelu_exact(acc0[r] + bias);
        Hbuf[(1 * kK2S + k2s) * 512 + slot] = (__bf16)gelu_exact(acc1[r] + bias);
      }
    }
  }
  __syncthreads();

  // ---- Phase 2: out = hidden @ W2 + b2 (f32 output) ----
  {
    const long bstep = (long)(kD >> 4) * 512;
    for (int ch = 0; ch < kD / 128; ++ch) {
      const int nt = ch * 8 + wave;
      const __bf16* bptr = W2f + (long)nt * 512 + lane * 16;
      v8f acc0 = {}, acc1 = {};
#pragma unroll 4
      for (int ks = 0; ks < kK2S; ++ks) {
        v16bf b  = *reinterpret_cast<const v16bf*>(bptr + (long)ks * bstep);
        v16bf a0 = *reinterpret_cast<const v16bf*>(
            Hbuf + (ks * 32 + lane) * 16);
        v16bf a1 = *reinterpret_cast<const v16bf*>(
            Hbuf + ((kK2S + ks) * 32 + lane) * 16);
        acc0 = __builtin_amdgcn_wmma_f32_16x16x32_bf16(false, a0, false, b,
                                                       (short)0, acc0, false, false);
        acc1 = __builtin_amdgcn_wmma_f32_16x16x32_bf16(false, a1, false, b,
                                                       (short)0, acc1, false, false);
      }
      const int col = nt * 16 + lcol;
      const float bias = B2[col];
#pragma unroll
      for (int r = 0; r < 8; ++r) {
        long grow0 = row0 + lrow + r;
        Out[row_addr(out_map, grow0)      + col] = acc0[r] + bias;
        Out[row_addr(out_map, grow0 + 16) + col] = acc1[r] + bias;
      }
    }
  }
}

// One block per (batch, class) token: gate logits via block reduction,
// exact reference top-1 semantics, blend the two expert outputs.
__global__ void gate_select_kernel(const float* __restrict__ X,
                                   const float* __restrict__ GW,
                                   const float* __restrict__ Ebuf,
                                   float* __restrict__ Out)
{
  const int token = blockIdx.x;           // 0..383
  const int b = token / kNcls;
  const int i = token % kNcls;
  const int t = threadIdx.x;

  const float* tok = X + ((long)b * (kNcls + kP) + i) * kD;
  float p0 = 0.f, p1 = 0.f;
  for (int d = t; d < kD; d += 256) {
    float xv = tok[d];
    p0 += xv * GW[((long)i * kD + d) * 2 + 0];
    p1 += xv * GW[((long)i * kD + d) * 2 + 1];
  }
  __shared__ float s0[256], s1[256];
  __shared__ float wsel[2];
  s0[t] = p0; s1[t] = p1;
  __syncthreads();
  for (int off = 128; off > 0; off >>= 1) {
    if (t < off) { s0[t] += s0[t + off]; s1[t] += s1[t + off]; }
    __syncthreads();
  }
  if (t == 0) {
    float l0 = s0[0], l1 = s1[0];
    float mx = fmaxf(l0, l1);
    float e0 = expf(l0 - mx), e1 = expf(l1 - mx);
    float inv = 1.f / (e0 + e1);
    float g0 = e0 * inv, g1 = e1 * inv;
    float thr = fminf(g0, g1);                      // sorted[-(K+1)], K=1, 2 gates
    float m0 = (g0 > thr) ? 1.f : 0.f;
    float m1 = (g1 > thr) ? 1.f : 0.f;
    float s = fmaxf(g0 * m0 + g1 * m1, 1e-6f);      // clip(sum, 1e-6)
    wsel[0] = g0 * m0 / s;
    wsel[1] = g1 * m1 / s;
  }
  __syncthreads();
  const float w0 = wsel[0], w1 = wsel[1];
  const float* E0 = Ebuf + (((long)i * 2 + 0) * 64 + b) * kD;
  const float* E1 = Ebuf + (((long)i * 2 + 1) * 64 + b) * kD;
  float* o = Out + ((long)b * (kNcls + kP) + i) * kD;
  for (int d = t; d < kD; d += 256)
    o[d] = w0 * E0[d] + w1 * E1[d];
}

extern "C" void kernel_launch(void* const* d_in, const int* in_sizes, int n_in,
                              void* d_out, int out_size, void* d_ws, size_t ws_size,
                              hipStream_t stream)
{
  (void)in_sizes; (void)n_in; (void)out_size; (void)ws_size;
  const float* x   = (const float*)d_in[0];
  const float* W1  = (const float*)d_in[1];
  const float* b1  = (const float*)d_in[2];
  const float* W2  = (const float*)d_in[3];
  const float* b2  = (const float*)d_in[4];
  const float* A1W = (const float*)d_in[5];
  const float* A1b = (const float*)d_in[6];
  const float* A2W = (const float*)d_in[7];
  const float* A2b = (const float*)d_in[8];
  const float* GW  = (const float*)d_in[9];
  // d_in[10] = mids : unused by the reference computation.
  float* out = (float*)d_out;

  // Workspace layout: expert outputs (f32) then fragment-packed bf16 weights.
  float*  ews = (float*)d_ws;                              // 12*64*768 f32
  __bf16* W1f = (__bf16*)((char*)d_ws + 12L * 64 * kD * sizeof(float));
  __bf16* W2f = W1f + kWElems;
  __bf16* A1f = W2f + kWElems;                             // 5 matrices
  __bf16* A2f = A1f + 5 * kWElems;

  // ---- Pack weights into WMMA fragment order (bf16) ----
  const int packBlocks = (int)((kWElems / 8 + 255) / 256);  // 1152
  pack_weights_kernel<<<packBlocks, 256, 0, stream>>>(W1, W1f, kD, kH);
  pack_weights_kernel<<<packBlocks, 256, 0, stream>>>(W2, W2f, kH, kD);
  for (int j = 0; j < 5; ++j) {
    pack_weights_kernel<<<packBlocks, 256, 0, stream>>>(
        A1W + (long)j * kWElems, A1f + (long)j * kWElems, kD, kH);
    pack_weights_kernel<<<packBlocks, 256, 0, stream>>>(
        A2W + (long)j * kWElems, A2f + (long)j * kWElems, kH, kD);
  }

  const size_t smem = (size_t)kSmemElems * sizeof(__bf16);  // 245760 B < 320 KB

  // Patch path: rows r -> x[r/256, 6 + r%256, :]
  RowMap pin  { (long)kNcls * kD, (long)(kNcls + kP) * kD, kD, kP };
  RowMap pout = pin;
  fused_mlp_kernel<<<dim3((64 * kP) / kBM, 1), 256, smem, stream>>>(
      x, W1f, b1, W2f, b2, out, pin, pout, 0, nullptr, nullptr, nullptr, nullptr);

  // Cls experts: 12 experts x 64 batch rows -> workspace
  RowMap cin  { 0, 0, (long)(kNcls + kP) * kD, 1L << 30 };
  RowMap cout { 0, 0, kD, 1L << 30 };
  fused_mlp_kernel<<<dim3(64 / kBM, 12), 256, smem, stream>>>(
      x, nullptr, nullptr, nullptr, nullptr, ews, cin, cout, 1,
      A1f, A1b, A2f, A2b);

  // Gating + blend into cls rows of the output
  gate_select_kernel<<<dim3(64 * kNcls), 256, 0, stream>>>(x, GW, ews, out);
}
